// TbNetV1_82489141887097
// MI455X (gfx1250) — matrix operand order repaired and checked by reading, not compiled, not run on hardware
//
#include <hip/hip_runtime.h>
#include <math.h>

// Problem constants (match reference)
#define N_NODES 25000
#define N_EDGES 400000
#define F_IN    8
#define HD      128
#define T_SEQ   20
#define EMBD    64

typedef __attribute__((ext_vector_type(16))) _Float16 v16h;
typedef __attribute__((ext_vector_type(8)))  _Float16 v8h;
typedef __attribute__((ext_vector_type(8)))  float    v8f;

// ---------------------------------------------------------------------------
// WMMA fragment helpers (CDNA5 v_wmma_f32_16x16x32_f16 layouts, wave32)
// ---------------------------------------------------------------------------

// A-matrix 16x32 f16 fragment from an f16 row-major LDS tile A[16][lda].
// Lane l: row = l&15; lanes 0-15 hold K = {0..7, 16..23}, lanes 16-31 hold
// K = {8..15, 24..31} (relative to k0): two aligned 16-byte LDS loads.
__device__ __forceinline__ v16h load_a_frag(const _Float16* A, int lda, int k0, int lane) {
    const int row = lane & 15;
    const int hi  = lane >> 4;             // 0 or 1
    const _Float16* r = A + row * lda;
    v8h lo = *(const v8h*)(r + k0 + hi * 8);
    v8h up = *(const v8h*)(r + k0 + 16 + hi * 8);
    v16h a;
#pragma unroll
    for (int e = 0; e < 8; ++e) { a[e] = lo[e]; a[8 + e] = up[e]; }
    return a;
}

// B-matrix 32x16 f16 fragment from a pre-swizzled buffer: tile (nt,kt) stored
// as 32 lanes x 16 contiguous halfs (one 32-byte load per lane).
__device__ __forceinline__ v16h load_b_frag(const _Float16* Bsw, int kTiles,
                                            int nt, int kt, int lane) {
    return *(const v16h*)(Bsw + ((size_t)(nt * kTiles + kt) * 32 + lane) * 16);
}

__device__ __forceinline__ v8f wmma_acc(v16h a, v16h b, v8f c) {
    return __builtin_amdgcn_wmma_f32_16x16x32_f16(
        /*neg_a=*/false, a, /*neg_b=*/false, b,
        /*c_mod=*/(short)0, c, /*reuse_a=*/false, /*reuse_b=*/false);
}

// ---------------------------------------------------------------------------
// Weight pre-swizzle: f32 row-major W[K][Ncols] -> f16 B-fragment layout.
// One thread per (tile, lane). tile = nt*kTiles + kt.
// ---------------------------------------------------------------------------
__global__ void tbnet_swizzle(const float* __restrict__ W, _Float16* __restrict__ out,
                              int K, int Ncols) {
    int kT = K >> 5, nT = Ncols >> 4;
    int idx = blockIdx.x * blockDim.x + threadIdx.x;
    int total = kT * nT * 32;
    if (idx >= total) return;
    int lane = idx & 31;
    int tile = idx >> 5;
    int kt = tile % kT;
    int nt = tile / kT;
    int col = (nt << 4) + (lane & 15);
    int kb  = (kt << 5) + ((lane >> 4) << 3);   // lanes 16-31 hold K+8..15, K+24..31
    _Float16* dst = out + ((size_t)tile * 32 + lane) * 16;
#pragma unroll
    for (int e = 0; e < 16; ++e)
        dst[e] = (_Float16)W[(size_t)(kb + (e & 7) + ((e >> 3) << 4)) * Ncols + col];
}

// ---------------------------------------------------------------------------
// GCN support kernels
// ---------------------------------------------------------------------------
__global__ void tbnet_fill(float* p, int n, float v) {
    int i = blockIdx.x * blockDim.x + threadIdx.x;
    if (i < n) p[i] = v;
}

__global__ void tbnet_degree(const int* __restrict__ dst, float* deg, int ne) {
    int i = blockIdx.x * blockDim.x + threadIdx.x;
    if (i < ne) atomicAdd(&deg[dst[i]], 1.0f);
}

__global__ void tbnet_dinv(const float* __restrict__ deg, float* dinv, int n) {
    int i = blockIdx.x * blockDim.x + threadIdx.x;
    if (i < n) dinv[i] = rsqrtf(fmaxf(deg[i], 1.0f));
}

// hw = x @ W1  (K=8: plain VALU, too small for WMMA)
__global__ void tbnet_gcn1_gemm(const float* __restrict__ x, const float* __restrict__ w1,
                                float* __restrict__ hw, int n) {
    int i = blockIdx.x * blockDim.x + threadIdx.x;
    if (i >= n * HD) return;
    int node = i >> 7, c = i & (HD - 1);
    float s = 0.f;
#pragma unroll
    for (int k = 0; k < F_IN; ++k) s += x[node * F_IN + k] * w1[k * HD + c];
    hw[i] = s;
}

// acc[i] = dinv[i]^2 * hw[i]  (self-loop term, norm = dinv*dinv)
__global__ void tbnet_self_init(const float* __restrict__ dinv, const float* __restrict__ hw,
                                float* __restrict__ acc, int n) {
    int i = blockIdx.x * blockDim.x + threadIdx.x;
    if (i >= n * HD) return;
    float d = dinv[i >> 7];
    acc[i] = d * d * hw[i];
}

// acc[dst] += dinv[src]*dinv[dst] * hw[src]; 32 threads/edge x 4 cols each
__global__ void tbnet_scatter(const int* __restrict__ e0, const int* __restrict__ e1,
                              const float* __restrict__ dinv, const float* __restrict__ hw,
                              float* __restrict__ acc, int ne) {
    long idx = (long)blockIdx.x * blockDim.x + threadIdx.x;
    if (idx >= (long)ne * 32) return;
    int edge = (int)(idx >> 5);
    int c0 = (int)(idx & 31) * 4;
    int s = e0[edge], d = e1[edge];
    float nrm = dinv[s] * dinv[d];
#pragma unroll
    for (int j = 0; j < 4; ++j)
        atomicAdd(&acc[(size_t)d * HD + c0 + j], nrm * hw[(size_t)s * HD + c0 + j]);
}

__global__ void tbnet_finalize(const float* __restrict__ acc, const float* __restrict__ b,
                               float* __restrict__ out, int n) {
    int i = blockIdx.x * blockDim.x + threadIdx.x;
    if (i >= n * HD) return;
    out[i] = fmaxf(acc[i] + b[i & (HD - 1)], 0.f);
}

// ---------------------------------------------------------------------------
// GCN layer-2 GEMM: Out[N,128] = A[N,128] @ Bsw  (WMMA, 8 waves = 8 col tiles)
// Block-uniform fast path for full M-tiles (all but the last block).
// ---------------------------------------------------------------------------
__global__ __launch_bounds__(256) void tbnet_gemm_h(const float* __restrict__ A,
                                                    const _Float16* __restrict__ Bsw,
                                                    float* __restrict__ Out, int n) {
    __shared__ _Float16 sA[16 * HD];       // 4 KB, f16 staged once
    int m0 = blockIdx.x << 4;
    int tid = threadIdx.x;
    const bool full = (m0 + 16 <= n);      // scalar: uniform across block
    if (full) {
        for (int e = tid; e < 16 * HD; e += 256)
            sA[e] = (_Float16)A[(size_t)m0 * HD + e];
    } else {
        for (int e = tid; e < 16 * HD; e += 256) {
            int m = m0 + (e >> 7);
            sA[e] = (m < n) ? (_Float16)A[(size_t)m * HD + (e & (HD - 1))] : (_Float16)0.f;
        }
    }
    __syncthreads();
    int lane = tid & 31, wave = tid >> 5;            // wave = column tile
    v8f acc = {};
#pragma unroll
    for (int kt = 0; kt < 4; ++kt)
        acc = wmma_acc(load_a_frag(sA, HD, kt << 5, lane),
                       load_b_frag(Bsw, 4, wave, kt, lane), acc);
    int col = (wave << 4) + (lane & 15);
    int rb  = (lane < 16) ? 0 : 8;
    if (full) {
#pragma unroll
        for (int r = 0; r < 8; ++r)
            Out[(size_t)(m0 + rb + r) * HD + col] = acc[r];
    } else {
#pragma unroll
        for (int r = 0; r < 8; ++r) {
            int m = m0 + rb + r;
            if (m < n) Out[(size_t)m * HD + col] = acc[r];
        }
    }
}

// ---------------------------------------------------------------------------
// GRU: one block = 16 nodes. Per step: xp = emb@W_ih+b, gh = h@W_hh+b (WMMA),
// then fused r/z/n gates. 24 column tiles -> 3 per wave. LDS = 54 KB.
// h kept in f16 (f32 accumulation inside WMMA).
// ---------------------------------------------------------------------------
__global__ __launch_bounds__(256) void tbnet_gru(const int* __restrict__ xtext,
                                                 const float* __restrict__ embed,
                                                 const _Float16* __restrict__ wih,
                                                 const _Float16* __restrict__ whh,
                                                 const float* __restrict__ bih,
                                                 const float* __restrict__ bhh,
                                                 float* __restrict__ txt, int n) {
    __shared__ _Float16 sEmb[16 * EMBD];   // 2 KB
    __shared__ float    sX[16 * 3 * HD];   // 24 KB
    __shared__ float    sG[16 * 3 * HD];   // 24 KB
    __shared__ _Float16 sH[16 * HD];       // 4 KB
    int m0 = blockIdx.x << 4;
    int tid = threadIdx.x, lane = tid & 31, wave = tid >> 5;
    const bool full = (m0 + 16 <= n);      // uniform

    for (int e = tid; e < 16 * HD; e += 256) sH[e] = (_Float16)0.f;

    for (int t = 0; t < T_SEQ; ++t) {
        // gather embeddings for this timestep (convert to f16 once)
        if (full) {
            for (int e = tid; e < 16 * EMBD; e += 256) {
                int tok = xtext[(size_t)(m0 + (e >> 6)) * T_SEQ + t];
                sEmb[e] = (_Float16)embed[(size_t)tok * EMBD + (e & (EMBD - 1))];
            }
        } else {
            for (int e = tid; e < 16 * EMBD; e += 256) {
                int m = m0 + (e >> 6);
                float v = 0.f;
                if (m < n) {
                    int tok = xtext[(size_t)m * T_SEQ + t];
                    v = embed[(size_t)tok * EMBD + (e & (EMBD - 1))];
                }
                sEmb[e] = (_Float16)v;
            }
        }
        __syncthreads();   // sEmb ready; previous step's sH writes visible

#pragma unroll
        for (int j = 0; j < 3; ++j) {
            int nt = wave * 3 + j;                  // 24 gate-column tiles
            v8f ax = {};
#pragma unroll
            for (int kt = 0; kt < 2; ++kt)          // K = EMB = 64
                ax = wmma_acc(load_a_frag(sEmb, EMBD, kt << 5, lane),
                              load_b_frag(wih, 2, nt, kt, lane), ax);
            v8f ag = {};
#pragma unroll
            for (int kt = 0; kt < 4; ++kt)          // K = H = 128
                ag = wmma_acc(load_a_frag(sH, HD, kt << 5, lane),
                              load_b_frag(whh, 4, nt, kt, lane), ag);
            int col = (nt << 4) + (lane & 15);
            int rb  = (lane < 16) ? 0 : 8;
            float bx = bih[col], bg = bhh[col];
#pragma unroll
            for (int r = 0; r < 8; ++r) {
                sX[(rb + r) * 3 * HD + col] = ax[r] + bx;
                sG[(rb + r) * 3 * HD + col] = ag[r] + bg;
            }
        }
        __syncthreads();

        // fused gate math: r,z,n ; h = (1-z)*n + z*h
        for (int e = tid; e < 16 * HD; e += 256) {
            int r = e >> 7, c = e & (HD - 1);
            const float* xr = &sX[r * 3 * HD];
            const float* gr = &sG[r * 3 * HD];
            float rg = 1.f / (1.f + expf(-(xr[c] + gr[c])));
            float zg = 1.f / (1.f + expf(-(xr[c + HD] + gr[c + HD])));
            float ng = tanhf(xr[c + 2 * HD] + rg * gr[c + 2 * HD]);
            float hprev = (float)sH[e];
            sH[e] = (_Float16)((1.f - zg) * ng + zg * hprev);
        }
        // next iteration's post-gather barrier orders sH writes vs GEMM reads
    }
    __syncthreads();
    if (full) {
        for (int e = tid; e < 16 * HD; e += 256)
            txt[(size_t)m0 * HD + e] = (float)sH[e];
    } else {
        for (int e = tid; e < 16 * HD; e += 256) {
            int m = m0 + (e >> 7);
            if (m < n) txt[(size_t)m * HD + (e & (HD - 1))] = (float)sH[e];
        }
    }
}

// ---------------------------------------------------------------------------
// Edge pipeline: 16 edges/block. Three pair-MLPs (K=256) into ef[16,384] f16,
// then two heads (K=384 -> relu -> K=128x2 dot -> log_softmax). All WMMA.
// LDS = 28 KB. E is a multiple of 16, so the full-tile fast path always runs.
// ---------------------------------------------------------------------------
__global__ __launch_bounds__(256) void tbnet_edge(
    const int* __restrict__ e0, const int* __restrict__ e1,
    const float* __restrict__ pos, const float* __restrict__ txtf,
    const float* __restrict__ img,
    const _Float16* __restrict__ wp, const float* __restrict__ bp,
    const _Float16* __restrict__ wt, const float* __restrict__ bt,
    const _Float16* __restrict__ wi, const float* __restrict__ bi,
    const _Float16* __restrict__ wr1, const float* __restrict__ br1,
    const float* __restrict__ wr2, const float* __restrict__ br2,
    const _Float16* __restrict__ wc1, const float* __restrict__ bc1,
    const float* __restrict__ wc2, const float* __restrict__ bc2,
    float* __restrict__ outr, float* __restrict__ outc, int ne) {
    __shared__ _Float16 sPair[16 * 2 * HD];   // 8 KB
    __shared__ _Float16 sEf[16 * 3 * HD];     // 12 KB
    __shared__ float    sHid[16 * HD];        // 8 KB
    int m0 = blockIdx.x << 4;
    int tid = threadIdx.x, lane = tid & 31, wave = tid >> 5;
    const bool full = (m0 + 16 <= ne);        // uniform; always true for E=400000

    for (int f = 0; f < 3; ++f) {
        const float* feat    = (f == 0) ? pos : (f == 1) ? txtf : img;
        const _Float16* wsw  = (f == 0) ? wp  : (f == 1) ? wt   : wi;
        const float* bias    = (f == 0) ? bp  : (f == 1) ? bt   : bi;
        if (full) {
            for (int e = tid; e < 16 * 2 * HD; e += 256) {
                int m = m0 + (e >> 8);
                int c = e & (2 * HD - 1);
                int node = (c < HD) ? e0[m] : e1[m];
                sPair[e] = (_Float16)feat[(size_t)node * HD + (c & (HD - 1))];
            }
        } else {
            for (int e = tid; e < 16 * 2 * HD; e += 256) {
                int m = m0 + (e >> 8);
                int c = e & (2 * HD - 1);
                float v = 0.f;
                if (m < ne) {
                    int node = (c < HD) ? e0[m] : e1[m];
                    v = feat[(size_t)node * HD + (c & (HD - 1))];
                }
                sPair[e] = (_Float16)v;
            }
        }
        __syncthreads();
        v8f acc = {};
#pragma unroll
        for (int kt = 0; kt < 8; ++kt)     // K = 256
            acc = wmma_acc(load_a_frag(sPair, 2 * HD, kt << 5, lane),
                           load_b_frag(wsw, 8, wave, kt, lane), acc);
        int col = (wave << 4) + (lane & 15);
        int rb  = (lane < 16) ? 0 : 8;
        float bb = bias[col];
#pragma unroll
        for (int r = 0; r < 8; ++r)
            sEf[(rb + r) * 3 * HD + f * HD + col] = (_Float16)fmaxf(acc[r] + bb, 0.f);
        __syncthreads();
    }

    for (int head = 0; head < 2; ++head) {
        const _Float16* w1 = head ? wc1 : wr1;
        const float* b1    = head ? bc1 : br1;
        const float* w2    = head ? wc2 : wr2;
        const float* b2    = head ? bc2 : br2;
        float* out         = head ? outc : outr;
        v8f acc = {};
#pragma unroll
        for (int kt = 0; kt < 12; ++kt)    // K = 384
            acc = wmma_acc(load_a_frag(sEf, 3 * HD, kt << 5, lane),
                           load_b_frag(w1, 12, wave, kt, lane), acc);
        int col = (wave << 4) + (lane & 15);
        int rb  = (lane < 16) ? 0 : 8;
        float bb = b1[col];
#pragma unroll
        for (int r = 0; r < 8; ++r)
            sHid[(rb + r) * HD + col] = fmaxf(acc[r] + bb, 0.f);
        __syncthreads();
        if (tid < 16) {
            int m = m0 + tid;
            if (full || m < ne) {
                float l0 = b2[0], l1 = b2[1];
                const float* hrow = &sHid[tid * HD];
#pragma unroll 4
                for (int k = 0; k < HD; ++k) {
                    float h = hrow[k];
                    l0 += h * w2[k * 2 + 0];
                    l1 += h * w2[k * 2 + 1];
                }
                float mx  = fmaxf(l0, l1);
                float lse = mx + logf(expf(l0 - mx) + expf(l1 - mx));
                out[(size_t)m * 2 + 0] = l0 - lse;
                out[(size_t)m * 2 + 1] = l1 - lse;
            }
        }
        __syncthreads();   // before col head reuses sHid
    }
}

// ---------------------------------------------------------------------------
extern "C" void kernel_launch(void* const* d_in, const int* in_sizes, int n_in,
                              void* d_out, int out_size, void* d_ws, size_t ws_size,
                              hipStream_t stream) {
    (void)in_sizes; (void)n_in; (void)out_size; (void)ws_size;
    const float* x      = (const float*)d_in[0];
    const float* img    = (const float*)d_in[1];
    const int*   eidx   = (const int*)d_in[2];
    const int*   xtext  = (const int*)d_in[3];
    const float* gcn_w1 = (const float*)d_in[4];
    const float* gcn_b1 = (const float*)d_in[5];
    const float* gcn_w2 = (const float*)d_in[6];
    const float* gcn_b2 = (const float*)d_in[7];
    const float* embed  = (const float*)d_in[8];
    const float* w_ih   = (const float*)d_in[9];
    const float* w_hh   = (const float*)d_in[10];
    const float* b_ih   = (const float*)d_in[11];
    const float* b_hh   = (const float*)d_in[12];
    const float* pos_w  = (const float*)d_in[13];
    const float* pos_b  = (const float*)d_in[14];
    const float* text_w = (const float*)d_in[15];
    const float* text_b = (const float*)d_in[16];
    const float* img_w  = (const float*)d_in[17];
    const float* img_b  = (const float*)d_in[18];
    const float* row_w1 = (const float*)d_in[19];
    const float* row_b1 = (const float*)d_in[20];
    const float* row_w2 = (const float*)d_in[21];
    const float* row_b2 = (const float*)d_in[22];
    const float* col_w1 = (const float*)d_in[23];
    const float* col_b1 = (const float*)d_in[24];
    const float* col_w2 = (const float*)d_in[25];
    const float* col_b2 = (const float*)d_in[26];

    const int* e0 = eidx;
    const int* e1 = eidx + N_EDGES;

    char* ws = (char*)d_ws;
    size_t off = 0;
    auto alloc = [&](size_t bytes) -> void* {
        void* p = ws + off;
        off += (bytes + 255) & ~(size_t)255;
        return p;
    };
    float* deg  = (float*)alloc((size_t)N_NODES * 4);
    float* dinv = (float*)alloc((size_t)N_NODES * 4);
    float* hw   = (float*)alloc((size_t)N_NODES * HD * 4);
    float* acc  = (float*)alloc((size_t)N_NODES * HD * 4);
    float* posb = (float*)alloc((size_t)N_NODES * HD * 4);
    float* txtb = (float*)alloc((size_t)N_NODES * HD * 4);
    _Float16* sw_gcn2 = (_Float16*)alloc((size_t)HD * HD * 2);
    _Float16* sw_wih  = (_Float16*)alloc((size_t)EMBD * 3 * HD * 2);
    _Float16* sw_whh  = (_Float16*)alloc((size_t)HD * 3 * HD * 2);
    _Float16* sw_pw   = (_Float16*)alloc((size_t)2 * HD * HD * 2);
    _Float16* sw_tw   = (_Float16*)alloc((size_t)2 * HD * HD * 2);
    _Float16* sw_iw   = (_Float16*)alloc((size_t)2 * HD * HD * 2);
    _Float16* sw_rw1  = (_Float16*)alloc((size_t)3 * HD * HD * 2);
    _Float16* sw_cw1  = (_Float16*)alloc((size_t)3 * HD * HD * 2);

    float* out_row = (float*)d_out;
    float* out_col = out_row + (size_t)N_EDGES * 2;

    dim3 blk(256);
    auto g1 = [](long n) { return dim3((unsigned)((n + 255) / 256)); };

    // Pre-swizzle all WMMA B operands (f32 -> f16 fragment layout)
    tbnet_swizzle<<<g1(4L * 8 * 32),  blk, 0, stream>>>(gcn_w2, sw_gcn2, HD, HD);
    tbnet_swizzle<<<g1(2L * 24 * 32), blk, 0, stream>>>(w_ih,  sw_wih, EMBD, 3 * HD);
    tbnet_swizzle<<<g1(4L * 24 * 32), blk, 0, stream>>>(w_hh,  sw_whh, HD,   3 * HD);
    tbnet_swizzle<<<g1(8L * 8 * 32),  blk, 0, stream>>>(pos_w,  sw_pw, 2 * HD, HD);
    tbnet_swizzle<<<g1(8L * 8 * 32),  blk, 0, stream>>>(text_w, sw_tw, 2 * HD, HD);
    tbnet_swizzle<<<g1(8L * 8 * 32),  blk, 0, stream>>>(img_w,  sw_iw, 2 * HD, HD);
    tbnet_swizzle<<<g1(12L * 8 * 32), blk, 0, stream>>>(row_w1, sw_rw1, 3 * HD, HD);
    tbnet_swizzle<<<g1(12L * 8 * 32), blk, 0, stream>>>(col_w1, sw_cw1, 3 * HD, HD);

    // Degrees (self-loop folded into init value 1.0)
    tbnet_fill  <<<g1(N_NODES), blk, 0, stream>>>(deg, N_NODES, 1.0f);
    tbnet_degree<<<g1(N_EDGES), blk, 0, stream>>>(e1, deg, N_EDGES);
    tbnet_dinv  <<<g1(N_NODES), blk, 0, stream>>>(deg, dinv, N_NODES);

    // GCN layer 1 (K=8 -> VALU GEMM, then normalized scatter-add)
    tbnet_gcn1_gemm<<<g1((long)N_NODES * HD), blk, 0, stream>>>(x, gcn_w1, hw, N_NODES);
    tbnet_self_init<<<g1((long)N_NODES * HD), blk, 0, stream>>>(dinv, hw, acc, N_NODES);
    tbnet_scatter  <<<g1((long)N_EDGES * 32), blk, 0, stream>>>(e0, e1, dinv, hw, acc, N_EDGES);
    tbnet_finalize <<<g1((long)N_NODES * HD), blk, 0, stream>>>(acc, gcn_b1, posb, N_NODES);

    // GCN layer 2 (WMMA GEMM)
    tbnet_gemm_h   <<<dim3((N_NODES + 15) / 16), blk, 0, stream>>>(posb, sw_gcn2, hw, N_NODES);
    tbnet_self_init<<<g1((long)N_NODES * HD), blk, 0, stream>>>(dinv, hw, acc, N_NODES);
    tbnet_scatter  <<<g1((long)N_EDGES * 32), blk, 0, stream>>>(e0, e1, dinv, hw, acc, N_EDGES);
    tbnet_finalize <<<g1((long)N_NODES * HD), blk, 0, stream>>>(acc, gcn_b2, posb, N_NODES);

    // GRU over text (WMMA for both input & recurrent projections)
    tbnet_gru<<<dim3((N_NODES + 15) / 16), blk, 0, stream>>>(
        xtext, embed, sw_wih, sw_whh, b_ih, b_hh, txtb, N_NODES);

    // Edge pair MLPs + heads + log_softmax (fused, all WMMA)
    tbnet_edge<<<dim3((N_EDGES + 15) / 16), blk, 0, stream>>>(
        e0, e1, posb, txtb, img,
        sw_pw, pos_b, sw_tw, text_b, sw_iw, img_b,
        sw_rw1, row_b1, row_w2, row_b2,
        sw_cw1, col_b1, col_w2, col_b2,
        out_row, out_col, N_EDGES);
}
